// E2EModel_36558761624076
// MI455X (gfx1250) — matrix operand order repaired and checked by reading, not compiled, not run on hardware
//
#include <hip/hip_runtime.h>
#include <hip/hip_bf16.h>
#include <math.h>

typedef __attribute__((ext_vector_type(16))) _Float16 v16h;
typedef __attribute__((ext_vector_type(8)))  float    v8f;
typedef int v4i_t __attribute__((vector_size(16)));

#define SRC_V 50000
#define TGT_V 32000
#define B_    64
#define S_    128
#define T_    64
#define D_    128   // = E = H = A
#define NBLK  (TGT_V / 128)   // 250 column blocks in the logits GEMM

// ---------------------------------------------------------------------------
// WMMA fragment helpers (layouts per CDNA5 ISA 7.12.2, wave32)
// ---------------------------------------------------------------------------
__device__ inline v16h frag_a(const _Float16* A, int lda, int m_base, int k_base, int lane) {
    int m = m_base + (lane & 15);
    int g = (lane >> 4) & 1;
    const _Float16* p = A + (size_t)m * lda + k_base + g * 8;
    union { v16h v; uint4 q[2]; } u;
    u.q[0] = *(const uint4*)(p);
    u.q[1] = *(const uint4*)(p + 16);
    return u.v;
}

__device__ inline v16h frag_b(const _Float16* W, int ldw, int n_base, int k_base, int lane) {
    int n = n_base + (lane & 15);
    int g = (lane >> 4) & 1;
    const _Float16* p = W + (size_t)n * ldw + k_base + g * 8;
    union { v16h v; uint4 q[2]; } u;
    u.q[0] = *(const uint4*)(p);
    u.q[1] = *(const uint4*)(p + 16);
    return u.v;
}

__device__ inline v8f wmma_step(v16h a, v16h b, v8f c) {
    return __builtin_amdgcn_wmma_f32_16x16x32_f16(false, a, false, b, (short)0, c,
                                                  false, false);
}

// C[16x16] tile of A[MxK] @ W[NxK]^T accumulated over K in steps of 32.
__device__ inline v8f wmma_kloop(const _Float16* A, int lda,
                                 const _Float16* W, int ldw,
                                 int m_base, int n_base, int K, int lane) {
    v8f c = {};
    for (int k = 0; k < K; k += 32)
        c = wmma_step(frag_a(A, lda, m_base, k, lane), frag_b(W, ldw, n_base, k, lane), c);
    return c;
}

// ---------------------------------------------------------------------------
// Async global->LDS tile copy (CDNA5 GLOBAL_LOAD_ASYNC_TO_LDS_B128, ASYNCcnt)
// Falls back to plain b128 copies if the builtin is unavailable.
// ---------------------------------------------------------------------------
#if __has_builtin(__builtin_amdgcn_global_load_async_to_lds_b128)
#define HAVE_ASYNC_LDS 1
#else
#define HAVE_ASYNC_LDS 0
#endif

typedef __attribute__((address_space(1))) v4i_t* g_v4i_p;
typedef __attribute__((address_space(3))) v4i_t* l_v4i_p;

__device__ inline void wait_async_lds() {
#if HAVE_ASYNC_LDS
#if __has_builtin(__builtin_amdgcn_s_wait_asynccnt)
    __builtin_amdgcn_s_wait_asynccnt(0);
#else
    asm volatile("s_wait_asynccnt 0" ::: "memory");
#endif
#endif
}

// Copies 16 KB (1024 x 16B chunks) with 256 threads.
__device__ inline void copy_tile16k(const _Float16* __restrict__ g, _Float16* l, int tid) {
#if HAVE_ASYNC_LDS
    for (int i = tid; i < 1024; i += 256)
        __builtin_amdgcn_global_load_async_to_lds_b128(
            (g_v4i_p)(g + i * 8), (l_v4i_p)(l + i * 8), 0, 0);
    wait_async_lds();
#else
    const uint4* s4 = (const uint4*)g;
    uint4* d4 = (uint4*)l;
    for (int i = tid; i < 1024; i += 256) d4[i] = s4[i];
#endif
}

// ---------------------------------------------------------------------------
// f32 -> f16 weight conversion (grid-stride)
// ---------------------------------------------------------------------------
__global__ void k_cvt(const float* __restrict__ src, _Float16* __restrict__ dst, int n) {
    for (int i = blockIdx.x * blockDim.x + threadIdx.x; i < n; i += gridDim.x * blockDim.x)
        dst[i] = (_Float16)src[i];
}

// ---------------------------------------------------------------------------
// Encoder: one block per source position s (128 blocks x 256 threads, 8 waves)
// ---------------------------------------------------------------------------
__global__ void k_encoder(const float* __restrict__ emb, const int* __restrict__ batch_x,
                          const _Float16* __restrict__ Wenc16, const float* __restrict__ b_enc,
                          const _Float16* __restrict__ Uw16, const float* __restrict__ U_b,
                          float* __restrict__ enc_out, float* __restrict__ uh) {
    __shared__ __align__(16) _Float16 Xs[B_ * D_];
    __shared__ __align__(16) _Float16 Eo[B_ * D_];
    const int s   = blockIdx.x;
    const int tid = threadIdx.x;

    for (int i = tid; i < B_ * D_; i += 256) {
        int b = i >> 7, d = i & 127;
        int tok = batch_x[s * B_ + b];
        Xs[i] = (_Float16)emb[(size_t)tok * D_ + d];
    }
    __syncthreads();

    const int wave = tid >> 5, lane = tid & 31;
    const int n = wave * 16 + (lane & 15);
    const int g = (lane >> 4) & 1;

    for (int mt = 0; mt < 4; ++mt) {
        v8f c = wmma_kloop(Xs, D_, Wenc16, D_, mt * 16, wave * 16, D_, lane);
        float bias = b_enc[n];
#pragma unroll
        for (int v = 0; v < 8; ++v) {
            int m = mt * 16 + v + 8 * g;
            float val = c[v] + bias;
            val = val > 0.0f ? val : 0.0f;
            enc_out[((size_t)s * B_ + m) * D_ + n] = val;
            Eo[m * D_ + n] = (_Float16)val;
        }
    }
    __syncthreads();

    for (int mt = 0; mt < 4; ++mt) {
        v8f c = wmma_kloop(Eo, D_, Uw16, D_, mt * 16, wave * 16, D_, lane);
        float bias = U_b[n];
#pragma unroll
        for (int v = 0; v < 8; ++v) {
            int m = mt * 16 + v + 8 * g;
            uh[((size_t)s * B_ + m) * D_ + n] = c[v] + bias;
        }
    }
}

// h0[b][e] = sum_s enc_out[s][b][e]
__global__ void k_h0(const float* __restrict__ enc_out, float* __restrict__ h_cur) {
    int i = blockIdx.x * blockDim.x + threadIdx.x;
    if (i < B_ * D_) {
        float acc = 0.0f;
        for (int s = 0; s < S_; ++s) acc += enc_out[(size_t)s * (B_ * D_) + i];
        h_cur[i] = acc;
    }
}

// ---------------------------------------------------------------------------
// Sequential decoder: single workgroup, 1024 threads = 32 waves.
// ---------------------------------------------------------------------------
__global__ void k_decoder(const float* __restrict__ emb, const int* __restrict__ batch_y,
                          const _Float16* __restrict__ Ww16, const float* __restrict__ Wb,
                          const float* __restrict__ v_w, const float* __restrict__ v_b,
                          const _Float16* __restrict__ Wc16, const float* __restrict__ Wc_b,
                          const _Float16* __restrict__ Wih16, const float* __restrict__ bih,
                          const _Float16* __restrict__ Whh16, const float* __restrict__ bhh,
                          const float* __restrict__ uh, const float* __restrict__ enc_out,
                          float* __restrict__ h_cur, float* __restrict__ wq,
                          _Float16* __restrict__ cat16, _Float16* __restrict__ rnn16,
                          float* __restrict__ gi, float* __restrict__ gh,
                          _Float16* __restrict__ h16_all) {
    __shared__ __align__(16) _Float16 hb[B_ * D_];  // 16 KB
    __shared__ float sc[S_ * B_];                   // 32 KB
    const int tid = threadIdx.x;
    const int wave = tid >> 5, lane = tid & 31;
    const float vb0 = v_b[0];

    for (int t = 0; t < T_; ++t) {
        for (int i = tid; i < B_ * D_; i += 1024) hb[i] = (_Float16)h_cur[i];
        for (int i = tid; i < B_ * D_; i += 1024) {
            int b = i >> 7, d = i & 127;
            int tok = (t == 0) ? (SRC_V - 1) : batch_y[(t - 1) * B_ + b];
            cat16[b * 256 + d] = (_Float16)emb[(size_t)tok * D_ + d];
        }
        __threadfence(); __syncthreads();

        // wq = hb @ Ww^T + Wb
        {
            int nt = wave & 7, mt = wave >> 3;
            v8f c = wmma_kloop(hb, D_, Ww16, D_, mt * 16, nt * 16, D_, lane);
            int n = nt * 16 + (lane & 15); int g = (lane >> 4) & 1;
            float bias = Wb[n];
#pragma unroll
            for (int v = 0; v < 8; ++v) wq[(mt * 16 + v + 8 * g) * D_ + n] = c[v] + bias;
        }
        __threadfence(); __syncthreads();

        // scores
        for (int i = tid; i < S_ * B_; i += 1024) {
            int s = i >> 6, b = i & 63;
            const float* uhp = uh + ((size_t)s * B_ + b) * D_;
            const float* wqp = wq + b * D_;
            float acc = vb0;
            for (int a = 0; a < D_; ++a) acc += v_w[a] * tanhf(wqp[a] + uhp[a]);
            sc[s * B_ + b] = acc;
        }
        __syncthreads();

        // softmax over s per batch column
        if (tid < B_) {
            int b = tid;
            float mx = -INFINITY;
            for (int s = 0; s < S_; ++s) mx = fmaxf(mx, sc[s * B_ + b]);
            float sum = 0.0f;
            for (int s = 0; s < S_; ++s) { float e = __expf(sc[s * B_ + b] - mx); sc[s * B_ + b] = e; sum += e; }
            float inv = 1.0f / sum;
            for (int s = 0; s < S_; ++s) sc[s * B_ + b] *= inv;
        }
        __syncthreads();

        // ctx -> cat[:,128:]
        for (int i = tid; i < B_ * D_; i += 1024) {
            int b = i >> 7, e = i & 127;
            float acc = 0.0f;
            for (int s = 0; s < S_; ++s)
                acc += sc[s * B_ + b] * enc_out[((size_t)s * B_ + b) * D_ + e];
            cat16[b * 256 + 128 + e] = (_Float16)acc;
        }
        __threadfence(); __syncthreads();

        // rnn_in = cat @ Wc^T + Wc_b  (K=256)
        {
            int nt = wave & 7, mt = wave >> 3;
            v8f c = wmma_kloop(cat16, 256, Wc16, 256, mt * 16, nt * 16, 256, lane);
            int n = nt * 16 + (lane & 15); int g = (lane >> 4) & 1;
            float bias = Wc_b[n];
#pragma unroll
            for (int v = 0; v < 8; ++v)
                rnn16[(mt * 16 + v + 8 * g) * D_ + n] = (_Float16)(c[v] + bias);
        }
        __threadfence(); __syncthreads();

        // gate GEMMs: gi (rnn16 @ Wih^T) and gh (hb @ Whh^T)
        for (int j = 0; j < 6; ++j) {
            int id = wave * 6 + j;
            bool is_gh = id >= 96;
            int id2 = is_gh ? id - 96 : id;
            int mt = id2 / 24, nt = id2 % 24;
            const _Float16* A = is_gh ? hb : rnn16;
            const _Float16* W = is_gh ? Whh16 : Wih16;
            const float* bias = is_gh ? bhh : bih;
            float* outp = is_gh ? gh : gi;
            v8f c = wmma_kloop(A, D_, W, D_, mt * 16, nt * 16, D_, lane);
            int n = nt * 16 + (lane & 15); int g = (lane >> 4) & 1;
            float bv = bias[n];
#pragma unroll
            for (int v = 0; v < 8; ++v)
                outp[(mt * 16 + v + 8 * g) * 384 + n] = c[v] + bv;
        }
        __threadfence(); __syncthreads();

        // GRU update
        for (int i = tid; i < B_ * D_; i += 1024) {
            int b = i >> 7, hx = i & 127;
            float ir = gi[b * 384 + hx],        hr = gh[b * 384 + hx];
            float iz = gi[b * 384 + 128 + hx],  hz = gh[b * 384 + 128 + hx];
            float in_ = gi[b * 384 + 256 + hx], hn = gh[b * 384 + 256 + hx];
            float r = 1.0f / (1.0f + __expf(-(ir + hr)));
            float z = 1.0f / (1.0f + __expf(-(iz + hz)));
            float ng = tanhf(in_ + r * hn);
            float hnew = (1.0f - z) * ng + z * h_cur[i];
            h_cur[i] = hnew;
            h16_all[(size_t)t * (B_ * D_) + i] = (_Float16)hnew;
        }
        __threadfence(); __syncthreads();
    }
}

// ---------------------------------------------------------------------------
// Logits pass 1: compute 64x128 logit tiles in LDS, emit per-(row, col-block)
// partial max and sum(exp) only. No 512MB logit write in this pass.
// Grid: (NBLK, 64) x 256 threads.
// ---------------------------------------------------------------------------
__global__ void k_logits_pass1(const _Float16* __restrict__ h16,
                               const _Float16* __restrict__ Wout16,
                               const float* __restrict__ bout,
                               float* __restrict__ pmax, float* __restrict__ psum) {
    __shared__ __align__(16) _Float16 As[64 * 128];  // 16 KB
    __shared__ float Ls[64 * 128];                   // 32 KB tile logits
    __shared__ float smax[64 * 4], ssum[64 * 4], rmax[64];
    const int bn = blockIdx.x, bm = blockIdx.y;
    const int tid = threadIdx.x;

    copy_tile16k(h16 + (size_t)bm * 64 * 128, As, tid);
    __syncthreads();

    const int wave = tid >> 5, lane = tid & 31;
    const int n_base = bn * 128 + wave * 16;
    const int nl = wave * 16 + (lane & 15);           // local column
    const int g = (lane >> 4) & 1;
    const float bias = bout[bn * 128 + nl];

    // Hoist B fragments (rows of Wout, all K) out of the M loop.
    v16h b0 = frag_b(Wout16, 128, n_base, 0,  lane);
    v16h b1 = frag_b(Wout16, 128, n_base, 32, lane);
    v16h b2 = frag_b(Wout16, 128, n_base, 64, lane);
    v16h b3 = frag_b(Wout16, 128, n_base, 96, lane);

    for (int mt = 0; mt < 4; ++mt) {
        v8f c = {};
        c = wmma_step(frag_a(As, 128, mt * 16, 0,  lane), b0, c);
        c = wmma_step(frag_a(As, 128, mt * 16, 32, lane), b1, c);
        c = wmma_step(frag_a(As, 128, mt * 16, 64, lane), b2, c);
        c = wmma_step(frag_a(As, 128, mt * 16, 96, lane), b3, c);
#pragma unroll
        for (int v = 0; v < 8; ++v)
            Ls[(mt * 16 + v + 8 * g) * 128 + nl] = c[v] + bias;
    }
    __syncthreads();

    // Per-row reduction over the 128 local columns (4 threads per row).
    const int row = tid >> 2, seg = (tid & 3) * 32;
    float m = -INFINITY;
    for (int j = 0; j < 32; ++j) m = fmaxf(m, Ls[row * 128 + seg + j]);
    smax[tid] = m;
    __syncthreads();
    if (tid < 64) {
        float mm = fmaxf(fmaxf(smax[tid * 4], smax[tid * 4 + 1]),
                         fmaxf(smax[tid * 4 + 2], smax[tid * 4 + 3]));
        rmax[tid] = mm;
    }
    __syncthreads();
    float mr = rmax[row];
    float s = 0.0f;
    for (int j = 0; j < 32; ++j) s += __expf(Ls[row * 128 + seg + j] - mr);
    ssum[tid] = s;
    __syncthreads();
    if (tid < 64) {
        float ss = ssum[tid * 4] + ssum[tid * 4 + 1] + ssum[tid * 4 + 2] + ssum[tid * 4 + 3];
        int grow = bm * 64 + tid;
        pmax[(size_t)grow * NBLK + bn] = rmax[tid];
        psum[(size_t)grow * NBLK + bn] = ss;
    }
}

// Merge NBLK partials per row into the row's log-sum-exp.
__global__ void k_lse_reduce(const float* __restrict__ pmax, const float* __restrict__ psum,
                             float* __restrict__ lse) {
    int r = blockIdx.x * blockDim.x + threadIdx.x;
    if (r < T_ * B_) {
        const float* pm = pmax + (size_t)r * NBLK;
        const float* ps = psum + (size_t)r * NBLK;
        float M = -INFINITY;
        for (int j = 0; j < NBLK; ++j) M = fmaxf(M, pm[j]);
        float S = 0.0f;
        for (int j = 0; j < NBLK; ++j) S += ps[j] * __expf(pm[j] - M);
        lse[r] = M + logf(S);
    }
}

// ---------------------------------------------------------------------------
// Logits pass 2: recompute the GEMM (cheaper than re-reading 512MB) and write
// normalized log-softmax directly: out = c + bias - lse[row]. Single 512MB write.
// ---------------------------------------------------------------------------
__global__ void k_logits_pass2(const _Float16* __restrict__ h16,
                               const _Float16* __restrict__ Wout16,
                               const float* __restrict__ bout,
                               const float* __restrict__ lse,
                               float* __restrict__ out) {
    __shared__ __align__(16) _Float16 As[64 * 128];
    __shared__ float lsed[64];
    const int bn = blockIdx.x, bm = blockIdx.y;
    const int tid = threadIdx.x;

    copy_tile16k(h16 + (size_t)bm * 64 * 128, As, tid);
    if (tid < 64) lsed[tid] = lse[bm * 64 + tid];
    __syncthreads();

    const int wave = tid >> 5, lane = tid & 31;
    const int n_base = bn * 128 + wave * 16;
    const int n = n_base + (lane & 15);
    const int g = (lane >> 4) & 1;
    const float bias = bout[n];

    v16h b0 = frag_b(Wout16, 128, n_base, 0,  lane);
    v16h b1 = frag_b(Wout16, 128, n_base, 32, lane);
    v16h b2 = frag_b(Wout16, 128, n_base, 64, lane);
    v16h b3 = frag_b(Wout16, 128, n_base, 96, lane);

    for (int mt = 0; mt < 4; ++mt) {
        v8f c = {};
        c = wmma_step(frag_a(As, 128, mt * 16, 0,  lane), b0, c);
        c = wmma_step(frag_a(As, 128, mt * 16, 32, lane), b1, c);
        c = wmma_step(frag_a(As, 128, mt * 16, 64, lane), b2, c);
        c = wmma_step(frag_a(As, 128, mt * 16, 96, lane), b3, c);
#pragma unroll
        for (int v = 0; v < 8; ++v) {
            int ml = mt * 16 + v + 8 * g;
            int m = bm * 64 + ml;
            out[(size_t)m * TGT_V + n] = c[v] + bias - lsed[ml];
        }
    }
}

// ---------------------------------------------------------------------------
extern "C" void kernel_launch(void* const* d_in, const int* in_sizes, int n_in,
                              void* d_out, int out_size, void* d_ws, size_t ws_size,
                              hipStream_t stream) {
    const float* emb     = (const float*)d_in[0];
    const float* W_enc   = (const float*)d_in[1];
    const float* b_enc   = (const float*)d_in[2];
    const float* U_w     = (const float*)d_in[3];
    const float* U_b     = (const float*)d_in[4];
    const float* Ww      = (const float*)d_in[5];
    const float* Wb      = (const float*)d_in[6];
    const float* v_w     = (const float*)d_in[7];
    const float* v_b     = (const float*)d_in[8];
    const float* Wc_w    = (const float*)d_in[9];
    const float* Wc_b    = (const float*)d_in[10];
    const float* Wih     = (const float*)d_in[11];
    const float* bih     = (const float*)d_in[12];
    const float* Whh     = (const float*)d_in[13];
    const float* bhh     = (const float*)d_in[14];
    const float* Wout    = (const float*)d_in[15];
    const float* bout    = (const float*)d_in[16];
    const int*   batch_x = (const int*)d_in[17];
    const int*   batch_y = (const int*)d_in[18];
    float* out = (float*)d_out;

    char* w = (char*)d_ws;
    auto alloc = [&](size_t bytes) -> char* {
        char* p = w; w += (bytes + 255) & ~(size_t)255; return p;
    };
    _Float16* Wenc16 = (_Float16*)alloc(128 * 128 * 2);
    _Float16* Uw16   = (_Float16*)alloc(128 * 128 * 2);
    _Float16* Ww16   = (_Float16*)alloc(128 * 128 * 2);
    _Float16* Wc16   = (_Float16*)alloc(128 * 256 * 2);
    _Float16* Wih16  = (_Float16*)alloc(384 * 128 * 2);
    _Float16* Whh16  = (_Float16*)alloc(384 * 128 * 2);
    _Float16* Wout16 = (_Float16*)alloc((size_t)TGT_V * 128 * 2);
    float* enc_out   = (float*)alloc((size_t)S_ * B_ * D_ * 4);
    float* uh        = (float*)alloc((size_t)S_ * B_ * D_ * 4);
    float* h_cur     = (float*)alloc(B_ * D_ * 4);
    float* wq        = (float*)alloc(B_ * D_ * 4);
    _Float16* cat16  = (_Float16*)alloc(B_ * 256 * 2);
    _Float16* rnn16  = (_Float16*)alloc(B_ * D_ * 2);
    float* gi        = (float*)alloc(B_ * 384 * 4);
    float* gh        = (float*)alloc(B_ * 384 * 4);
    _Float16* h16    = (_Float16*)alloc((size_t)T_ * B_ * D_ * 2);
    float* pmax      = (float*)alloc((size_t)T_ * B_ * NBLK * 4);
    float* psum      = (float*)alloc((size_t)T_ * B_ * NBLK * 4);
    float* lse       = (float*)alloc((size_t)T_ * B_ * 4);

    // Weight conversions
    k_cvt<<<64, 256, 0, stream>>>(W_enc, Wenc16, 128 * 128);
    k_cvt<<<64, 256, 0, stream>>>(U_w,   Uw16,   128 * 128);
    k_cvt<<<64, 256, 0, stream>>>(Ww,    Ww16,   128 * 128);
    k_cvt<<<128, 256, 0, stream>>>(Wc_w, Wc16,   128 * 256);
    k_cvt<<<192, 256, 0, stream>>>(Wih,  Wih16,  384 * 128);
    k_cvt<<<192, 256, 0, stream>>>(Whh,  Whh16,  384 * 128);
    k_cvt<<<4096, 256, 0, stream>>>(Wout, Wout16, TGT_V * 128);

    // Encoder + initial hidden state
    k_encoder<<<S_, 256, 0, stream>>>(emb, batch_x, Wenc16, b_enc, Uw16, U_b, enc_out, uh);
    k_h0<<<(B_ * D_) / 256, 256, 0, stream>>>(enc_out, h_cur);

    // Sequential GRU decoder (single WGP, 32 waves)
    k_decoder<<<1, 1024, 0, stream>>>(emb, batch_y, Ww16, Wb, v_w, v_b,
                                      Wc16, Wc_b, Wih16, bih, Whh16, bhh,
                                      uh, enc_out, h_cur, wq, cat16, rnn16,
                                      gi, gh, h16);

    // Fused log-softmax: partials -> LSE -> recompute + single normalized write
    k_logits_pass1<<<dim3(NBLK, (T_ * B_) / 64), 256, 0, stream>>>(h16, Wout16, bout, pmax, psum);
    k_lse_reduce<<<(T_ * B_ + 255) / 256, 256, 0, stream>>>(pmax, psum, lse);
    k_logits_pass2<<<dim3(NBLK, (T_ * B_) / 64), 256, 0, stream>>>(h16, Wout16, bout, lse, out);
}
